// DeepHedgingNetwork_61452392071932
// MI455X (gfx1250) — matrix operand order repaired and checked by reading, not compiled, not run on hardware
//
#include <hip/hip_runtime.h>
#include <hip/hip_bf16.h>
#include <math.h>

typedef __attribute__((ext_vector_type(16))) _Float16 v16h;
typedef __attribute__((ext_vector_type(8)))  _Float16 v8h;
typedef __attribute__((ext_vector_type(8)))  float    v8f;
typedef __attribute__((ext_vector_type(4)))  float    v4f;
typedef __attribute__((ext_vector_type(2)))  float    v2f;

#define B_TOT   32768
#define T_STEPS 60
#define F_IN    5
#define H_DIM   256
#define BT      64      // batch rows per workgroup
#define NTHREADS 128    // 4 waves (wave32)
#define NWAVES  4

union Frag16 { v16h v; v8h h[2]; };

static __device__ __forceinline__ v8f wmma_f16(v16h a, v16h b, v8f c) {
  return __builtin_amdgcn_wmma_f32_16x16x32_f16(false, a, false, b, (short)0, c, false, false);
}

static __device__ __forceinline__ v8f wmma_f32k4(v2f a, v2f b, v8f c) {
  return __builtin_amdgcn_wmma_f32_16x16x4_f32(false, a, false, b, (short)0, c, false, false);
}

static __device__ __forceinline__ float relu1(float x) {
  return __builtin_amdgcn_fmed3f(x, 0.0f, __builtin_inff());
}

// ---- prep: W2T[n][k] = (f16) W2[k][n], 256x256 (A-operand for layer 2) ----
__global__ __launch_bounds__(256) void w2t_kernel(const float* __restrict__ W2,
                                                  _Float16* __restrict__ W2T) {
  int idx = blockIdx.x * 256 + threadIdx.x;   // 0..65535
  int n = idx >> 8, k = idx & 255;
  W2T[n * H_DIM + k] = (_Float16)W2[k * H_DIM + n];
}

__global__ __launch_bounds__(NTHREADS) void hedge_kernel(
    const float* __restrict__ feats,       // (B,T,F)
    const float* __restrict__ S,           // (B,T)
    const float* __restrict__ delta_init,  // (B)
    const float* __restrict__ W1,          // (8,256) in-dim major
    const float* __restrict__ b1,
    const float* __restrict__ b2,
    const _Float16* __restrict__ W2T,      // (256 n, 256 k) f16
    const float* __restrict__ W3,          // (256)
    const float* __restrict__ b3,          // (1)
    const float* __restrict__ y_in,        // (1)
    float* __restrict__ out,               // (B,T) deltas [+ y]
    int out_size)
{
  // x^T as f32 B-operand rows: xbuf[m][k], k = 0..7
  __shared__ float xbuf[BT][8];
  // W1^T as f32 A-operand: W1A[j][k], k = 0..7
  __shared__ float W1A[H_DIM][8];
  // layer-1 activations, batch-row major: h1s[m][j] (f16 B-operand for layer 2)
  __shared__ _Float16 h1s[BT][H_DIM];
  __shared__ float b1s[H_DIM], b2s[H_DIM], w3s[H_DIM];
  __shared__ float dacc2[NWAVES][BT];      // per-wave partial deltas (no atomics)

  const int tid  = threadIdx.x;
  const int lane = tid & 31;
  const int wave = tid >> 5;
  const int lan16 = lane & 15;
  const bool hi   = lane >= 16;
  const int b0 = blockIdx.x * BT;

  // ---------------- one-time init ----------------
  for (int i = tid; i < H_DIM; i += NTHREADS) {
    b1s[i] = b1[i]; b2s[i] = b2[i]; w3s[i] = W3[i];
  }
  for (int i = tid; i < H_DIM * 8; i += NTHREADS) {
    int j = i >> 3, k = i & 7;
    W1A[j][k] = W1[k * H_DIM + j];
  }

  // recurrent state lives entirely in registers of threads 0..63
  float st_dprev = 0.0f, st_pnl = 0.0f, st_vol = 0.0f;
  float f0 = 0.f, f1 = 0.f, f2 = 0.f, f3 = 0.f, f4 = 0.f;  // features(t)
  float s_cur = 0.0f, s_prev = 0.0f;                        // S(t), S(t-1)
  const float* my_feats = nullptr;
  const float* my_S = nullptr;
  if (tid < BT) {
    const int b = b0 + tid;
    st_dprev = delta_init[b];
    my_feats = feats + (size_t)b * T_STEPS * F_IN;
    my_S     = S + (size_t)b * T_STEPS;
    f0 = my_feats[0]; f1 = my_feats[1]; f2 = my_feats[2];
    f3 = my_feats[3]; f4 = my_feats[4];
    s_cur = my_S[0]; s_prev = s_cur;    // dS(0) = 0
  }
  __syncthreads();

  // ---------------- sequential time scan ----------------
  for (int t = 0; t < T_STEPS; ++t) {
    // 1) x = [feat(5), delta_prev, pnl, vol]  (recurrent feats BEFORE update)
    if (tid < BT) {
      xbuf[tid][0] = f0;
      xbuf[tid][1] = f1;
      xbuf[tid][2] = f2;
      xbuf[tid][3] = f3;
      xbuf[tid][4] = f4;
      xbuf[tid][5] = st_dprev;
      xbuf[tid][6] = st_pnl;
      xbuf[tid][7] = st_vol;
    }
    __syncthreads();

    // software-pipelined prefetch of step t+1 inputs (latency hidden by WMMA)
    float nf0 = 0.f, nf1 = 0.f, nf2 = 0.f, nf3 = 0.f, nf4 = 0.f, ns = 0.f;
    if (tid < BT) {
      const int tn = (t + 1 < T_STEPS) ? t + 1 : t;
      const float* fn = my_feats + (size_t)tn * F_IN;
      nf0 = fn[0]; nf1 = fn[1]; nf2 = fn[2]; nf3 = fn[3]; nf4 = fn[4];
      ns = my_S[tn];
    }

    // 2) layer 1 (transposed, f32 K=4 WMMA x2): D[j][m] = W1^T @ x^T.
    {
      v2f bx0[4], bx1[4];
      #pragma unroll
      for (int mt = 0; mt < 4; ++mt) {
        const float* xp = &xbuf[mt * 16 + lan16][hi ? 2 : 0];
        bx0[mt] = *(const v2f*)xp;            // K pass 0: k0 = 0
        bx1[mt] = *(const v2f*)(xp + 4);      // K pass 1: k0 = 4
      }
      for (int jt = wave * 4; jt < wave * 4 + 4; ++jt) {
        const int j = jt * 16 + lan16;
        const float* wp = &W1A[j][hi ? 2 : 0];
        const v2f a0 = *(const v2f*)wp;       // K 0..3 block
        const v2f a1 = *(const v2f*)(wp + 4); // K 4..7 block
        const int jb = jt * 16 + (hi ? 8 : 0);
        const v4f b1lo = *(const v4f*)&b1s[jb];
        const v4f b1hi = *(const v4f*)&b1s[jb + 4];
        #pragma unroll
        for (int mt = 0; mt < 4; ++mt) {
          v8f c = {};
          c = wmma_f32k4(a0, bx0[mt], c);
          c = wmma_f32k4(a1, bx1[mt], c);
          v8h pack;
          #pragma unroll
          for (int i = 0; i < 8; ++i) {
            const float bias = (i < 4) ? b1lo[i] : b1hi[i - 4];
            pack[i] = (_Float16)relu1(c[i] + bias);
          }
          *(v8h*)&h1s[mt * 16 + lan16][jb] = pack;   // ONE b128 store
        }
      }
    }
    __syncthreads();

    // 3) layer 2 (transposed) + fused layer 3.
    float dpart[4] = {0.0f, 0.0f, 0.0f, 0.0f};
    for (int ng = 0; ng < 2; ++ng) {
      const int nt0 = wave * 4 + ng * 2;
      v8f acc[2][4];
      #pragma unroll
      for (int q = 0; q < 2; ++q)
        #pragma unroll
        for (int mt = 0; mt < 4; ++mt)
          acc[q][mt] = (v8f){};
      for (int kt = 0; kt < 8; ++kt) {
        Frag16 bfr[4];                          // B-frags shared by both ntiles
        #pragma unroll
        for (int mt = 0; mt < 4; ++mt) {
          const _Float16* lb = &h1s[mt * 16 + lan16][kt * 32 + (hi ? 16 : 0)];
          bfr[mt].h[0] = *(const v8h*)lb;
          bfr[mt].h[1] = *(const v8h*)(lb + 8);
        }
        #pragma unroll
        for (int q = 0; q < 2; ++q) {
          const int n = (nt0 + q) * 16 + lan16;
          const _Float16* ga = W2T + (size_t)n * H_DIM + kt * 32 + (hi ? 16 : 0);
          Frag16 afr;
          afr.h[0] = *(const v8h*)ga;           // K base..base+7
          afr.h[1] = *(const v8h*)(ga + 8);     // K base+8..base+15
          #pragma unroll
          for (int mt = 0; mt < 4; ++mt)
            acc[q][mt] = wmma_f16(afr.v, bfr[mt].v, acc[q][mt]);
        }
      }
      // epilogue: bias + relu + W3 dot over this lane's 8 hidden units
      #pragma unroll
      for (int q = 0; q < 2; ++q) {
        const int nb = (nt0 + q) * 16 + (hi ? 8 : 0);
        const v4f b2lo = *(const v4f*)&b2s[nb];
        const v4f b2hi = *(const v4f*)&b2s[nb + 4];
        const v4f w3lo = *(const v4f*)&w3s[nb];
        const v4f w3hi = *(const v4f*)&w3s[nb + 4];
        #pragma unroll
        for (int mt = 0; mt < 4; ++mt) {
          float p = 0.0f;
          #pragma unroll
          for (int i = 0; i < 8; ++i) {
            const float bias = (i < 4) ? b2lo[i] : b2hi[i - 4];
            const float w3n  = (i < 4) ? w3lo[i] : w3hi[i - 4];
            p += relu1(acc[q][mt][i] + bias) * w3n;
          }
          dpart[mt] += p;
        }
      }
    }
    // merge half-row partials (lanes L and L+16 hold the same batch column m)
    #pragma unroll
    for (int mt = 0; mt < 4; ++mt) {
      float v = dpart[mt];
      v += __shfl_xor(v, 16, 32);
      if (!hi) dacc2[wave][mt * 16 + lan16] = v;   // plain store, no atomics
    }
    __syncthreads();

    // 4) recurrent state update in registers (matches reference ordering)
    if (tid < BT) {
      const float delta = dacc2[0][tid] + dacc2[1][tid] +
                          dacc2[2][tid] + dacc2[3][tid] + b3[0];
      const float ds = (t > 0) ? (s_cur - s_prev) : 0.0f;
      st_pnl += st_dprev * ds;
      st_vol += fabsf(delta - st_dprev);
      st_dprev = delta;
      out[(size_t)(b0 + tid) * T_STEPS + t] = delta;
      // shift the pipeline
      s_prev = s_cur; s_cur = ns;
      f0 = nf0; f1 = nf1; f2 = nf2; f3 = nf3; f4 = nf4;
    }
    __syncthreads();
  }

  if (blockIdx.x == 0 && tid == 0 && out_size > B_TOT * T_STEPS)
    out[(size_t)B_TOT * T_STEPS] = y_in[0];
}

extern "C" void kernel_launch(void* const* d_in, const int* in_sizes, int n_in,
                              void* d_out, int out_size, void* d_ws, size_t ws_size,
                              hipStream_t stream) {
  const float* feats      = (const float*)d_in[0];
  const float* S          = (const float*)d_in[1];
  const float* delta_init = (const float*)d_in[2];
  const float* W1         = (const float*)d_in[3];
  const float* b1         = (const float*)d_in[4];
  const float* W2         = (const float*)d_in[5];
  const float* b2         = (const float*)d_in[6];
  const float* W3         = (const float*)d_in[7];
  const float* b3         = (const float*)d_in[8];
  const float* y          = (const float*)d_in[9];
  float* out = (float*)d_out;

  _Float16* W2T = (_Float16*)d_ws;   // 256*256*2 = 128 KB of workspace

  w2t_kernel<<<H_DIM, 256, 0, stream>>>(W2, W2T);
  hedge_kernel<<<B_TOT / BT, NTHREADS, 0, stream>>>(
      feats, S, delta_init, W1, b1, b2, W2T, W3, b3, y, out, out_size);
}
